// MultiHeadSelfAttention_46789373723024
// MI455X (gfx1250) — matrix operand order repaired and checked by reading, not compiled, not run on hardware
//
#include <hip/hip_runtime.h>
#include <stdint.h>

#define D_DIM 1024
#define H_NUM 16
#define DH_DIM 64
#define B_NUM 4
#define T_NUM 2048
#define M_ROWS (B_NUM * T_NUM)

typedef __bf16 v16bf __attribute__((ext_vector_type(16)));
typedef float  v8f   __attribute__((ext_vector_type(8)));

union FragAB {
    v16bf v;
    uint4 q[2];
};

__device__ __forceinline__ unsigned short f32_to_bf16(float f) {
    union { float f; uint32_t u; } x;
    x.f = f;
    uint32_t r = 0x7FFFu + ((x.u >> 16) & 1u);
    return (unsigned short)((x.u + r) >> 16);
}

__device__ __forceinline__ uint32_t pk_bf16(float lo, float hi) {
    union { float f; uint32_t u; } a, b;
    a.f = lo;
    b.f = hi;
    uint32_t ra = (a.u + (0x7FFFu + ((a.u >> 16) & 1u))) >> 16;
    uint32_t rb = (b.u + (0x7FFFu + ((b.u >> 16) & 1u))) >> 16;
    return (rb << 16) | (ra & 0xFFFFu);
}

// ---- CDNA5 async copy: HBM -> LDS, tracked by ASYNCcnt -------------------
// GVS mode: mem_addr = SADDR(64b sgpr) + VADDR(32b vgpr) ; LDS addr in VDST.
__device__ __forceinline__ void async_load_b128(uint32_t lds_addr,
                                                uint32_t gvoff,
                                                uint64_t sbase) {
    asm volatile("global_load_async_to_lds_b128 %0, %1, %2"
                 :
                 : "v"(lds_addr), "v"(gvoff), "s"(sbase)
                 : "memory");
}
template <int N>
__device__ __forceinline__ void wait_asynccnt() {
    asm volatile("s_wait_asynccnt %0" ::"i"(N) : "memory");
}

// Butterfly reductions across the 16 lanes of a half-wave via ds_swizzle
__device__ __forceinline__ float red_max16(float v) {
    v = fmaxf(v, __int_as_float(__builtin_amdgcn_ds_swizzle(__float_as_int(v), 0x041f)));
    v = fmaxf(v, __int_as_float(__builtin_amdgcn_ds_swizzle(__float_as_int(v), 0x081f)));
    v = fmaxf(v, __int_as_float(__builtin_amdgcn_ds_swizzle(__float_as_int(v), 0x101f)));
    v = fmaxf(v, __int_as_float(__builtin_amdgcn_ds_swizzle(__float_as_int(v), 0x201f)));
    return v;
}
__device__ __forceinline__ float red_sum16(float v) {
    v += __int_as_float(__builtin_amdgcn_ds_swizzle(__float_as_int(v), 0x041f));
    v += __int_as_float(__builtin_amdgcn_ds_swizzle(__float_as_int(v), 0x081f));
    v += __int_as_float(__builtin_amdgcn_ds_swizzle(__float_as_int(v), 0x101f));
    v += __int_as_float(__builtin_amdgcn_ds_swizzle(__float_as_int(v), 0x201f));
    return v;
}

// ---------------------------------------------------------------------------
// One-time fp32 -> bf16 conversion (vectorized 4 elems/thread).
// ---------------------------------------------------------------------------
__global__ __launch_bounds__(256) void cvt_f32_bf16(const float* __restrict__ src,
                                                    unsigned short* __restrict__ dst,
                                                    int n4) {
    int i = blockIdx.x * blockDim.x + threadIdx.x;
    if (i < n4) {
        float4 f = ((const float4*)src)[i];
        uint2 o;
        o.x               = pk_bf16(f.x, f.y);
        o.y               = pk_bf16(f.z, f.w);
        ((uint2*)dst)[i]  = o;
    }
}

// ---------------------------------------------------------------------------
// Y[m,n] = sum_k A[m,k] * W[n,k] + bias[n]   (A, W already bf16)
// Block: 256 threads, tile 128(M) x 128(N), BK=32.
// Double-buffered LDS fed by global_load_async_to_lds_b128 (ASYNCcnt pipe).
// 8 waves as 4(M) x 2(N); each wave computes 32x64 (2x4 WMMA tiles).
// ---------------------------------------------------------------------------
template <bool OUT_IS_BF16>
__global__ __launch_bounds__(256) void gemm_wT_bias(
    const unsigned short* __restrict__ A, const unsigned short* __restrict__ W,
    const float* __restrict__ bias, void* __restrict__ Yptr) {
    __shared__ unsigned short As[2][128][40];  // row stride 80B (16B aligned)
    __shared__ unsigned short Bs[2][128][40];
    constexpr uint32_t BUF_SZ = 128 * 40 * 2;  // 10240 B per buffer

    const int tid  = threadIdx.x;
    const int wave = tid >> 5;
    const int lane = tid & 31;
    const int l16  = lane & 15;
    const int half = lane >> 4;
    const int wm   = wave & 3;   // 32-row M sub-tile
    const int wn   = wave >> 2;  // 64-col N sub-tile
    const int m0   = blockIdx.y * 128;
    const int n0   = blockIdx.x * 128;

    const uint32_t asBase = (uint32_t)(uintptr_t)&As[0][0][0];
    const uint32_t bsBase = (uint32_t)(uintptr_t)&Bs[0][0][0];
    const uint64_t aBase  = (uint64_t)(uintptr_t)A;
    const uint64_t wBase  = (uint64_t)(uintptr_t)W;

    // 128x32 halves = 512 16B-chunks, 2 per thread: chunk = pass*256 + tid
    auto issueTile = [&](int par, int k0) {
        uint64_t sa = aBase + ((size_t)m0 * D_DIM + k0) * 2;
        uint64_t sw = wBase + ((size_t)n0 * D_DIM + k0) * 2;
#pragma unroll
        for (int pass = 0; pass < 2; ++pass) {
            int chunk   = pass * 256 + tid;
            int row     = chunk >> 2;         // 0..127
            int cc      = (chunk & 3) * 8;    // 0,8,16,24 halves
            uint32_t ld = (uint32_t)(par * BUF_SZ + row * 80 + cc * 2);
            uint32_t go = (uint32_t)((row * D_DIM + cc) * 2);
            async_load_b128(asBase + ld, go, sa);
            async_load_b128(bsBase + ld, go, sw);
        }
    };

    v8f acc[2][4];
#pragma unroll
    for (int mt = 0; mt < 2; ++mt)
#pragma unroll
        for (int nt = 0; nt < 4; ++nt) acc[mt][nt] = {};

    issueTile(0, 0);

    int p = 0;
    for (int k0 = 0; k0 < D_DIM; k0 += 32, p ^= 1) {
        if (k0 + 32 < D_DIM) {
            issueTile(p ^ 1, k0 + 32);
            wait_asynccnt<4>();  // in-order: current tile's 4 ops are done
        } else {
            wait_asynccnt<0>();
        }
        __syncthreads();  // current-tile data visible to all waves

        FragAB a[2], b[4];
#pragma unroll
        for (int mt = 0; mt < 2; ++mt) {
            const unsigned short* pa = &As[p][wm * 32 + mt * 16 + l16][half * 8];
            a[mt].q[0] = *(const uint4*)pa;         // K = half*8 .. +7
            a[mt].q[1] = *(const uint4*)(pa + 16);  // K = 16+half*8 .. +7
        }
#pragma unroll
        for (int nt = 0; nt < 4; ++nt) {
            const unsigned short* pb = &Bs[p][wn * 64 + nt * 16 + l16][half * 8];
            b[nt].q[0] = *(const uint4*)pb;        // K = half*8 .. +7
            b[nt].q[1] = *(const uint4*)(pb + 16);  // K = 16+half*8 .. +7
        }
#pragma unroll
        for (int mt = 0; mt < 2; ++mt)
#pragma unroll
            for (int nt = 0; nt < 4; ++nt)
                acc[mt][nt] = __builtin_amdgcn_wmma_f32_16x16x32_bf16(
                    false, a[mt].v, false, b[nt].v, (short)0, acc[mt][nt],
                    false, false);

        __syncthreads();  // all reads of buffer p done -> safe to overwrite
    }

    // epilogue: +bias, convert, store
#pragma unroll
    for (int mt = 0; mt < 2; ++mt)
#pragma unroll
        for (int nt = 0; nt < 4; ++nt) {
            int n    = n0 + wn * 64 + nt * 16 + l16;
            float bv = bias[n];
#pragma unroll
            for (int r = 0; r < 8; ++r) {
                int m     = m0 + wm * 32 + mt * 16 + r + half * 8;
                float val = acc[mt][nt][r] + bv;
                if constexpr (OUT_IS_BF16)
                    ((unsigned short*)Yptr)[(size_t)m * D_DIM + n] = f32_to_bf16(val);
                else
                    ((float*)Yptr)[(size_t)m * D_DIM + n] = val;
            }
        }
}

// ---------------------------------------------------------------------------
// Fused flash attention: one block per (b, h, 128-query tile).
// K tile: async-to-LDS double buffer. V tile: register transpose (single buf).
// 8 waves, each owns 16 query rows. Online softmax, bf16 WMMA.
// ---------------------------------------------------------------------------
__global__ __launch_bounds__(256) void attention_fused(
    const unsigned short* __restrict__ Q, const unsigned short* __restrict__ K,
    const unsigned short* __restrict__ V, unsigned short* __restrict__ Ctx) {
    __shared__ unsigned short Kt[2][64][72];  // [key][dh]   stride 144B
    __shared__ unsigned short Vt[64][72];     // [dh][key]   (transposed)
    __shared__ unsigned short Pt[8][16][72];  // per-wave P  [qrow][key]
    constexpr uint32_t KBUF_SZ = 64 * 72 * 2;  // 9216 B

    const int tid  = threadIdx.x;
    const int wave = tid >> 5;
    const int lane = tid & 31;
    const int l16  = lane & 15;
    const int half = lane >> 4;

    const int qt = blockIdx.x;
    const int h  = blockIdx.y;
    const int b  = blockIdx.z;
    const int q0 = qt * 128;

    const size_t kvBase   = (size_t)(b * T_NUM) * D_DIM + h * DH_DIM;
    const uint32_t ktBase = (uint32_t)(uintptr_t)&Kt[0][0][0];
    const uint64_t kBase  = (uint64_t)(uintptr_t)K + kvBase * 2;

    // K: 64x64 halves = 512 16B-chunks, 2 per thread
    auto issueK = [&](int par, int kt) {
        uint64_t sk = kBase + (size_t)kt * D_DIM * 2;
#pragma unroll
        for (int pass = 0; pass < 2; ++pass) {
            int chunk = pass * 256 + tid;
            int key   = chunk >> 3;        // 0..63
            int cc    = (chunk & 7) * 8;   // 0..56 halves
            async_load_b128(ktBase + (uint32_t)(par * KBUF_SZ + key * 144 + cc * 2),
                            (uint32_t)((key * D_DIM + cc) * 2), sk);
        }
    };

    // V: pair of key rows per thread, 8 dh each -> 8 ds_store_b32 transposed
    const int kp  = tid >> 3;        // key pair id (0..31)
    const int d0V = (tid & 7) * 8;   // 8 dh per thread
    uint4 vReg[2];
    auto loadV = [&](int kt) {
        const unsigned short* vs = V + kvBase + (size_t)(kt + 2 * kp) * D_DIM + d0V;
        vReg[0] = *(const uint4*)vs;
        vReg[1] = *(const uint4*)(vs + D_DIM);
    };
    auto storeV = [&]() {
        const unsigned short* e0 = (const unsigned short*)&vReg[0];
        const unsigned short* e1 = (const unsigned short*)&vReg[1];
#pragma unroll
        for (int i = 0; i < 8; ++i) {
            uint32_t pair = (uint32_t)e0[i] | ((uint32_t)e1[i] << 16);
            *(uint32_t*)&Vt[d0V + i][2 * kp] = pair;  // [dh][key..key+1]
        }
    };

    // Q fragments (A-matrix, 16 rows x 64 dh = two K=32 chunks), loaded once.
    FragAB aq[2];
    {
        const size_t rowQ =
            (size_t)(b * T_NUM + q0 + wave * 16 + l16) * D_DIM + h * DH_DIM;
#pragma unroll
        for (int kc = 0; kc < 2; ++kc) {
            const unsigned short* pq = Q + rowQ + kc * 32 + half * 8;
            aq[kc].q[0] = *(const uint4*)pq;
            aq[kc].q[1] = *(const uint4*)(pq + 16);
        }
    }

    v8f acc[4] = {{}, {}, {}, {}};
    float mrun[8], lrun[8];
#pragma unroll
    for (int r = 0; r < 8; ++r) {
        mrun[r] = -1e30f;
        lrun[r] = 0.f;
    }

    issueK(0, 0);
    loadV(0);

    int p = 0;
    for (int kt = 0; kt < T_NUM; kt += 64, p ^= 1) {
        if (kt + 64 < T_NUM) {
            issueK(p ^ 1, kt + 64);
            wait_asynccnt<2>();  // current K tile's 2 ops done (in-order)
        } else {
            wait_asynccnt<0>();
        }
        storeV();         // Vt reads of previous tile finished at last barrier
        __syncthreads();  // K async data + Vt stores visible to all waves
        if (kt + 64 < T_NUM) loadV(kt + 64);  // HBM latency under compute
        if (kt + 128 < T_NUM)
            __builtin_prefetch(
                (const char*)(V + kvBase + (size_t)(kt + 128 + 2 * kp) * D_DIM + d0V),
                0, 1);

        // ---- S = Q * K^T : 16 x 64 per wave ----
        v8f s[4];
#pragma unroll
        for (int nt = 0; nt < 4; ++nt) {
            v8f c = {};
#pragma unroll
            for (int kc = 0; kc < 2; ++kc) {
                FragAB bk;  // B[k=dh][n=key] = K[key][dh]
                const unsigned short* pk = &Kt[p][nt * 16 + l16][kc * 32 + half * 8];
                bk.q[0] = *(const uint4*)pk;
                bk.q[1] = *(const uint4*)(pk + 16);
                c = __builtin_amdgcn_wmma_f32_16x16x32_bf16(
                    false, aq[kc].v, false, bk.v, (short)0, c, false, false);
            }
            s[nt] = c;
        }

        // ---- online softmax (row r + 8*half; 16 cols per lane-group) ----
#pragma unroll
        for (int r = 0; r < 8; ++r) {
            float mx = fmaxf(fmaxf(s[0][r], s[1][r]), fmaxf(s[2][r], s[3][r]));
            mx       = red_max16(mx * 0.125f);
            float mnew = fmaxf(mrun[r], mx);
            float corr = __expf(mrun[r] - mnew);
            float rs   = 0.f;
            unsigned short pb[4];
#pragma unroll
            for (int nt = 0; nt < 4; ++nt) {
                float pv = __expf(s[nt][r] * 0.125f - mnew);
                rs += pv;
                pb[nt] = f32_to_bf16(pv);
            }
            rs      = red_sum16(rs);
            lrun[r] = lrun[r] * corr + rs;
            mrun[r] = mnew;
#pragma unroll
            for (int dt = 0; dt < 4; ++dt) acc[dt][r] *= corr;
            int qrow = r + half * 8;
#pragma unroll
            for (int nt = 0; nt < 4; ++nt)
                Pt[wave][qrow][nt * 16 + l16] = pb[nt];
        }

        // ---- acc += P * V  (P via LDS to get A-fragment layout) ----
        FragAB ap[2];
#pragma unroll
        for (int kc = 0; kc < 2; ++kc) {
            const unsigned short* pp = &Pt[wave][l16][kc * 32 + half * 8];
            ap[kc].q[0] = *(const uint4*)pp;
            ap[kc].q[1] = *(const uint4*)(pp + 16);
        }
#pragma unroll
        for (int dt = 0; dt < 4; ++dt) {
#pragma unroll
            for (int kc = 0; kc < 2; ++kc) {
                FragAB bv;  // B[k=key][n=dh] = V[key][dh] -> Vt[dh][key]
                const unsigned short* pv = &Vt[dt * 16 + l16][kc * 32 + half * 8];
                bv.q[0] = *(const uint4*)pv;
                bv.q[1] = *(const uint4*)(pv + 16);
                acc[dt] = __builtin_amdgcn_wmma_f32_16x16x32_bf16(
                    false, ap[kc].v, false, bv.v, (short)0, acc[dt], false, false);
            }
        }

        __syncthreads();  // all reads of Kt[p] / Vt done -> safe to overwrite
    }

    // ---- finalize: ctx = acc / l, store bf16 ----
#pragma unroll
    for (int r = 0; r < 8; ++r) {
        float inv = 1.0f / lrun[r];
        int m     = b * T_NUM + q0 + wave * 16 + r + half * 8;
#pragma unroll
        for (int dt = 0; dt < 4; ++dt)
            Ctx[(size_t)m * D_DIM + h * DH_DIM + dt * 16 + l16] =
                f32_to_bf16(acc[dt][r] * inv);
    }
}

// ---------------------------------------------------------------------------
extern "C" void kernel_launch(void* const* d_in, const int* in_sizes, int n_in,
                              void* d_out, int out_size, void* d_ws,
                              size_t ws_size, hipStream_t stream) {
    (void)in_sizes; (void)n_in; (void)out_size; (void)ws_size;
    const float* x  = (const float*)d_in[0];
    const float* Wq = (const float*)d_in[1];
    const float* bq = (const float*)d_in[2];
    const float* Wk = (const float*)d_in[3];
    const float* bk = (const float*)d_in[4];
    const float* Wv = (const float*)d_in[5];
    const float* bv = (const float*)d_in[6];
    const float* Wo = (const float*)d_in[7];
    const float* bo = (const float*)d_in[8];
    float* out = (float*)d_out;

    // workspace (halves): xbf 8M | Wq/Wk/Wv/Wo 1M each | Q,K,V,Ctx 8M each
    unsigned short* xbf = (unsigned short*)d_ws;
    unsigned short* Wqb = xbf + (size_t)M_ROWS * D_DIM;
    unsigned short* Wkb = Wqb + (size_t)D_DIM * D_DIM;
    unsigned short* Wvb = Wkb + (size_t)D_DIM * D_DIM;
    unsigned short* Wob = Wvb + (size_t)D_DIM * D_DIM;
    unsigned short* Qbf = Wob + (size_t)D_DIM * D_DIM;
    unsigned short* Kbf = Qbf + (size_t)M_ROWS * D_DIM;
    unsigned short* Vbf = Kbf + (size_t)M_ROWS * D_DIM;
    unsigned short* Cbf = Vbf + (size_t)M_ROWS * D_DIM;

    dim3 blk(256);
    // one-time conversions
    int n4x = M_ROWS * D_DIM / 4;
    int n4w = D_DIM * D_DIM / 4;
    cvt_f32_bf16<<<(n4x + 255) / 256, blk, 0, stream>>>(x, xbf, n4x);
    cvt_f32_bf16<<<(n4w + 255) / 256, blk, 0, stream>>>(Wq, Wqb, n4w);
    cvt_f32_bf16<<<(n4w + 255) / 256, blk, 0, stream>>>(Wk, Wkb, n4w);
    cvt_f32_bf16<<<(n4w + 255) / 256, blk, 0, stream>>>(Wv, Wvb, n4w);
    cvt_f32_bf16<<<(n4w + 255) / 256, blk, 0, stream>>>(Wo, Wob, n4w);

    dim3 gGemm(D_DIM / 128, M_ROWS / 128);  // (8, 64)
    gemm_wT_bias<true><<<gGemm, blk, 0, stream>>>(xbf, Wqb, bq, Qbf);
    gemm_wT_bias<true><<<gGemm, blk, 0, stream>>>(xbf, Wkb, bk, Kbf);
    gemm_wT_bias<true><<<gGemm, blk, 0, stream>>>(xbf, Wvb, bv, Vbf);

    dim3 gAtt(T_NUM / 128, H_NUM, B_NUM);  // (16, 16, 4)
    attention_fused<<<gAtt, blk, 0, stream>>>(Qbf, Kbf, Vbf, Cbf);

    gemm_wT_bias<false><<<gGemm, blk, 0, stream>>>(Cbf, Wob, bo, out);
}
